// clustering_dynamic_learning_common_center_2_45286135169476
// MI455X (gfx1250) — matrix operand order repaired and checked by reading, not compiled, not run on hardware
//
#include <hip/hip_runtime.h>
#include <math.h>

typedef __attribute__((ext_vector_type(16))) _Float16     v16h;
typedef __attribute__((ext_vector_type(8)))  float        v8f;
typedef __attribute__((ext_vector_type(4)))  float        v4f;
typedef __attribute__((ext_vector_type(4)))  unsigned int v4u;
typedef __attribute__((ext_vector_type(8)))  int          v8i;
typedef __attribute__((ext_vector_type(4)))  int          v4i;

#define B_   32
#define N_   512
#define K_   32
#define T_   24
#define F_   64
#define C_   8
#define D_   16
#define MID_ 32
#define S_   64

#if defined(__has_builtin)
#if __has_builtin(__builtin_amdgcn_tensor_load_to_lds) && \
    __has_builtin(__builtin_amdgcn_s_wait_tensorcnt)
#define HAVE_TDM 1
#endif
#endif

// ---------------- Kernel A: centroid feature MLP (tiny, 1 block) -------------
__global__ void k_cenfeat(const float* __restrict__ cen,
                          const float* __restrict__ W1, const float* __restrict__ b1,
                          const float* __restrict__ W2, const float* __restrict__ b2,
                          const float* __restrict__ W3, const float* __restrict__ b3,
                          float* __restrict__ ws_cenf, float* __restrict__ ws_cnorm) {
  __shared__ float hid[C_][MID_];
  __shared__ float cf[C_][D_];
  int tid = threadIdx.x;
  if (tid < C_ * MID_) {
    int c = tid / MID_, m = tid % MID_;
    float acc = b1[m];
    for (int f = 0; f < F_; ++f) acc += cen[c * F_ + f] * W1[f * MID_ + m];
    hid[c][m] = fmaxf(acc, 0.f);
  }
  __syncthreads();
  if (tid < C_ * D_) {
    int c = tid / D_, d = tid % D_;
    float acc = b2[d];
    for (int m = 0; m < MID_; ++m) acc += hid[c][m] * W2[m * D_ + d];
    float acc2 = b3[d];
    for (int f = 0; f < F_; ++f) acc2 += cen[c * F_ + f] * W3[f * D_ + d];
    float v = fmaxf(acc, 0.f) + fmaxf(acc2, 0.f);
    cf[c][d] = v;
    ws_cenf[c * D_ + d] = v;
  }
  __syncthreads();
  if (tid < C_) {
    float s = 0.f;
    for (int d = 0; d < D_; ++d) s += cf[tid][d] * cf[tid][d];
    ws_cnorm[tid] = s;
  }
}

// ------------- Kernel B: per-node feature MLP + cdist + softmax --------------
__global__ __launch_bounds__(256) void k_simi(
    const float* __restrict__ x,
    const float* __restrict__ W1, const float* __restrict__ b1,
    const float* __restrict__ W2, const float* __restrict__ b2,
    const float* __restrict__ W3, const float* __restrict__ b3,
    const float* __restrict__ cenf, const float* __restrict__ cnorm,
    float* __restrict__ simi) {
  __shared__ float sW1[F_ * MID_];
  __shared__ float sb1[MID_];
  __shared__ float sW2[MID_ * D_];
  __shared__ float sb2[D_];
  __shared__ float sW3[F_ * D_];
  __shared__ float sb3[D_];
  __shared__ float scf[C_ * D_];
  __shared__ float scn[C_];
  int tid = threadIdx.x;
  for (int e = tid; e < F_ * MID_; e += 256) sW1[e] = W1[e];
  for (int e = tid; e < MID_ * D_; e += 256) sW2[e] = W2[e];
  for (int e = tid; e < F_ * D_;  e += 256) sW3[e] = W3[e];
  if (tid < MID_) sb1[tid] = b1[tid];
  if (tid < D_) { sb2[tid] = b2[tid]; sb3[tid] = b3[tid]; }
  if (tid < C_ * D_) scf[tid] = cenf[tid];
  if (tid < C_) scn[tid] = cnorm[tid];
  __syncthreads();

  int row = blockIdx.x * 256 + tid;  // 0 .. B*N-1
  float xr[F_];
  #pragma unroll 8
  for (int f = 0; f < F_; ++f) xr[f] = x[(size_t)row * F_ + f];
  float hid[MID_];
  for (int m = 0; m < MID_; ++m) {
    float a = sb1[m];
    #pragma unroll 8
    for (int f = 0; f < F_; ++f) a += xr[f] * sW1[f * MID_ + m];
    hid[m] = fmaxf(a, 0.f);
  }
  float ft[D_];
  float xn = 0.f;
  for (int d = 0; d < D_; ++d) {
    float a = sb2[d];
    #pragma unroll 8
    for (int m = 0; m < MID_; ++m) a += hid[m] * sW2[m * D_ + d];
    float a2 = sb3[d];
    #pragma unroll 8
    for (int f = 0; f < F_; ++f) a2 += xr[f] * sW3[f * D_ + d];
    float v = fmaxf(a, 0.f) + fmaxf(a2, 0.f);
    ft[d] = v;
    xn += v * v;
  }
  float dist[C_];
  float mx = -1e30f;
  #pragma unroll
  for (int c = 0; c < C_; ++c) {
    float dot = 0.f;
    #pragma unroll
    for (int d = 0; d < D_; ++d) dot += ft[d] * scf[c * D_ + d];
    float r = xn + scn[c] - 2.f * dot;
    r = sqrtf(fmaxf(r, 1e-30f));
    dist[c] = r;
    mx = fmaxf(mx, r);
  }
  float se = 0.f;
  #pragma unroll
  for (int c = 0; c < C_; ++c) { dist[c] = expf(dist[c] - mx); se += dist[c]; }
  float inv = 1.f / se;
  #pragma unroll
  for (int c = 0; c < C_; ++c) simi[(size_t)row * C_ + c] = dist[c] * inv;
}

// ---------------- Kernel C: gather + per-centroid GCN via WMMA ---------------
__global__ __launch_bounds__(256, 1) void k_gcn(
    const float* __restrict__ inp,   // (B,N,T)
    const int*   __restrict__ adj,   // (B,N,K)
    const float* __restrict__ gW,    // (C,T,S)
    const float* __restrict__ gBias, // (C,S)
    const float* __restrict__ simi,  // (B,N,C)
    float* __restrict__ out) {       // (B,N,C,S)
  __shared__ v16h  Wf[C_ * 4 * 32];          // 32 KB: B fragments [i][nt][lane]
  __shared__ float bLds[C_ * S_];            // 2 KB (filled via TDM if avail)
  __shared__ float wtile[8][C_][K_];         // 8 KB: per-wave simi, [i][k] order
  __shared__ int   idxL[8][K_];              // 1 KB

  int tid = threadIdx.x;
  int wid = tid >> 5, lane = tid & 31;

  // --- bias preload: Tensor Data Mover (1D, 512 x f32 -> LDS), wave0 only ---
#ifdef HAVE_TDM
  if (wid == 0) {
    unsigned long long ga = (unsigned long long)(uintptr_t)gBias;
    unsigned int lds = (unsigned int)(uintptr_t)&bLds[0];
    v4u g0 = {};
    g0[0] = 1u;                                   // count=1, user mode
    g0[1] = lds;                                  // lds_addr
    g0[2] = (unsigned int)ga;                     // global_addr[31:0]
    g0[3] = (unsigned int)((ga >> 32) & 0x01FFFFFFu) | (2u << 30); // addr hi | type=2
    v8i g1 = {};
    g1[0] = (int)(2u << 16);                      // data_size = 4B
    g1[1] = (int)(512u << 16);                    // tensor_dim0 = 512 (lo16 in [63:48])
    g1[2] = (int)(1u << 16);                      // tensor_dim1 = 1
    g1[3] = (int)(512u << 16);                    // tile_dim0 = 512
    g1[4] = 1;                                    // tile_dim1 = 1
    g1[5] = 512;                                  // tensor_dim0_stride
    v4i gz = {};
#if __clang_major__ >= 23
    v8i z8 = {};
    __builtin_amdgcn_tensor_load_to_lds(g0, g1, gz, gz, z8, 0);
#else
    __builtin_amdgcn_tensor_load_to_lds(g0, g1, gz, gz, 0);
#endif
    __builtin_amdgcn_s_wait_tensorcnt((short)0);
  }
#else
  for (int e = tid; e < C_ * S_; e += 256) bLds[e] = gBias[e];
#endif

  // --- pre-swizzle gcn_W into WMMA B-fragment layout (f16, T padded 24->32).
  // B tile (32x16): lane = col s%16 (lanes 16-31 repeat cols), half h ->
  // K = (h&7) + (lane>=16 ? 8 : 0) + (h>=8 ? 16 : 0).
  for (int e = tid; e < C_ * 4 * 32 * 16; e += 256) {
    int h  = e & 15;
    int ln = (e >> 4) & 31;
    int nt = (e >> 9) & 3;
    int i  = e >> 11;
    int t = (h & 7) + ((ln >= 16) ? 8 : 0) + ((h >= 8) ? 16 : 0);
    int s = nt * 16 + (ln & 15);
    float v = (t < T_) ? gW[(i * T_ + t) * S_ + s] : 0.f;
    ((_Float16*)&Wf[(i * 4 + nt) * 32 + ln])[h] = (_Float16)v;
  }
  __syncthreads();

  int gwave = blockIdx.x * 8 + wid;       // 4096 waves total
  for (int q = 0; q < 4; ++q) {
    int pair = gwave * 4 + q;             // 0 .. 16383
    int b = pair >> 9;                    // pair / N_
    // Gather neighbor indices + softmax weights (vectorized 2xb128 row load).
    int myidx = adj[(size_t)pair * K_ + lane];
    idxL[wid][lane] = myidx;
    const v4f* sp = (const v4f*)(simi + ((size_t)b * N_ + myidx) * C_);
    v4f s0 = sp[0], s1 = sp[1];
    float wreg[C_];
    #pragma unroll
    for (int i = 0; i < 4; ++i) { wreg[i] = s0[i]; wreg[i + 4] = s1[i]; }
    #pragma unroll
    for (int i = 0; i < C_; ++i) wtile[wid][i][lane] = wreg[i];
    // Butterfly-reduce over the 32 lanes -> all lanes hold wsum[i].
    #pragma unroll
    for (int st = 1; st < 32; st <<= 1)
      #pragma unroll
      for (int i = 0; i < C_; ++i) wreg[i] += __shfl_xor(wreg[i], st, 32);
    float invw[C_];
    #pragma unroll
    for (int i = 0; i < C_; ++i) invw[i] = 1.f / wreg[i];
    __syncthreads();

    // Build A fragments with b128 loads (rows are 96B => 16B aligned).
    // A layout: lane = row k%16; h<8 -> t = h + (lane>=16?8:0);
    // h>=8 -> t = h+8 (lane<16 only; lanes>=16 would be t>=24 -> stay 0).
    v16h a0 = {}, a1 = {};
    {
      int r0 = lane & 15;
      const v4f* rp0 = (const v4f*)(inp + ((size_t)b * N_ + idxL[wid][r0]) * T_);
      const v4f* rp1 = (const v4f*)(inp + ((size_t)b * N_ + idxL[wid][r0 + 16]) * T_);
      int qi = (lane >= 16) ? 2 : 0;
      v4f q0a = rp0[qi], q0b = rp0[qi + 1];
      v4f q1a = rp1[qi], q1b = rp1[qi + 1];
      #pragma unroll
      for (int h = 0; h < 4; ++h) {
        a0[h]     = (_Float16)q0a[h];
        a0[h + 4] = (_Float16)q0b[h];
        a1[h]     = (_Float16)q1a[h];
        a1[h + 4] = (_Float16)q1b[h];
      }
      if (lane < 16) {
        v4f q0c = rp0[4], q0d = rp0[5];
        v4f q1c = rp1[4], q1d = rp1[5];
        #pragma unroll
        for (int h = 0; h < 4; ++h) {
          a0[h + 8]  = (_Float16)q0c[h];
          a0[h + 12] = (_Float16)q0d[h];
          a1[h + 8]  = (_Float16)q1c[h];
          a1[h + 12] = (_Float16)q1d[h];
        }
      }
    }

    #pragma unroll 1
    for (int i = 0; i < C_; ++i) {
      // D layout: lane -> col s, VGPR r -> row r + (lane>=16?8:0).
      // wtile[i][k] contiguous in k -> 4 x ds_load_b128 for the 16 weights.
      int koff = (lane >= 16) ? 8 : 0;
      const v4f* wp = (const v4f*)&wtile[wid][i][0];
      v4f w0a = wp[koff >> 2], w0b = wp[(koff >> 2) + 1];
      v4f w1a = wp[(16 + koff) >> 2], w1b = wp[((16 + koff) >> 2) + 1];
      float wk0[8], wk1[8];
      #pragma unroll
      for (int r = 0; r < 4; ++r) {
        wk0[r] = w0a[r]; wk0[r + 4] = w0b[r];
        wk1[r] = w1a[r]; wk1[r + 4] = w1b[r];
      }
      float vals[4];
      #pragma unroll 2
      for (int nt = 0; nt < 4; ++nt) {
        v16h bfrag = Wf[(i * 4 + nt) * 32 + lane];
        int s = nt * 16 + (lane & 15);
        float bias = bLds[i * S_ + s];
        v8f cb;
        #pragma unroll
        for (int r = 0; r < 8; ++r) cb[r] = bias;   // bias folded into C
        v8f d0 = __builtin_amdgcn_wmma_f32_16x16x32_f16(
            false, a0, false, bfrag, (short)0, cb, false, false);
        v8f d1 = __builtin_amdgcn_wmma_f32_16x16x32_f16(
            false, a1, false, bfrag, (short)0, cb, false, false);
        float accA = 0.f, accB = 0.f;
        #pragma unroll
        for (int r = 0; r < 8; ++r) {
          accA += fmaxf(d0[r], 0.f) * wk0[r];
          accB += fmaxf(d1[r], 0.f) * wk1[r];
        }
        float acc = accA + accB;
        acc += __shfl_xor(acc, 16, 32);   // fold rows held by lane^16
        vals[nt] = acc * invw[i];
      }
      if (lane < 16) {
        size_t ob = ((size_t)pair * C_ + i) * S_ + lane;
        #pragma unroll
        for (int nt = 0; nt < 4; ++nt) out[ob + nt * 16] = vals[nt];
      }
    }
    __syncthreads();
  }
}

// ---------------- Kernels D/E/F: mean reduce + centroid update + loss --------
__global__ void k_zero(float* __restrict__ acc) { acc[threadIdx.x] = 0.f; }

__global__ __launch_bounds__(256) void k_reduce(const float* __restrict__ upd,
                                                float* __restrict__ acc) {
  int tid = threadIdx.x;
  int row0 = blockIdx.x * 16;
  float p0 = 0.f, p1 = 0.f;
  for (int r = 0; r < 16; ++r) {
    const float* rp = upd + (size_t)(row0 + r) * (C_ * S_);
    p0 += rp[tid];
    p1 += rp[tid + 256];
  }
  atomicAdd(&acc[tid], p0);
  atomicAdd(&acc[tid + 256], p1);
}

__global__ void k_final(const float* __restrict__ cen,
                        const float* __restrict__ acc,
                        float* __restrict__ out_loss) {
  __shared__ float nc[C_ * S_];
  __shared__ float red[64];
  int tid = threadIdx.x;
  if (tid < C_ * S_)
    nc[tid] = 0.99f * cen[tid] + 0.01f * (acc[tid] * (1.f / (B_ * N_)));
  __syncthreads();
  if (tid < 64) {
    int i = tid >> 3, j = tid & 7;
    float d2 = 0.f;
    for (int s = 0; s < S_; ++s) {
      float df = nc[i * S_ + s] - nc[j * S_ + s];
      d2 += df * df;
    }
    float d = sqrtf(fmaxf(d2, 1e-30f));
    float targ = (i == j) ? 0.f : 0.5f;
    float v = fmaxf(targ - d, 0.f);
    red[tid] = v * v;
  }
  __syncthreads();
  if (tid == 0) {
    float s = 0.f;
    for (int k = 0; k < 64; ++k) s += red[k];
    out_loss[0] = s;
  }
}

extern "C" void kernel_launch(void* const* d_in, const int* in_sizes, int n_in,
                              void* d_out, int out_size, void* d_ws, size_t ws_size,
                              hipStream_t stream) {
  const float* x    = (const float*)d_in[0];
  const float* inp  = (const float*)d_in[1];
  const int*   adj  = (const int*)d_in[2];
  const float* cen  = (const float*)d_in[3];
  const float* c1W1 = (const float*)d_in[4];
  const float* c1b1 = (const float*)d_in[5];
  const float* c1W2 = (const float*)d_in[6];
  const float* c1b2 = (const float*)d_in[7];
  const float* c2W  = (const float*)d_in[8];
  const float* c2b  = (const float*)d_in[9];
  const float* i1W1 = (const float*)d_in[10];
  const float* i1b1 = (const float*)d_in[11];
  const float* i1W2 = (const float*)d_in[12];
  const float* i1b2 = (const float*)d_in[13];
  const float* i2W  = (const float*)d_in[14];
  const float* i2b  = (const float*)d_in[15];
  const float* gW   = (const float*)d_in[16];
  const float* gB   = (const float*)d_in[17];

  float* out = (float*)d_out;
  float* ws  = (float*)d_ws;
  float* ws_cenf  = ws;                                // 128
  float* ws_cnorm = ws + 128;                          // 8
  float* ws_simi  = ws + 256;                          // B*N*C = 131072
  float* ws_acc   = ws + 256 + (size_t)B_ * N_ * C_;   // 512

  k_cenfeat<<<1, 256, 0, stream>>>(cen, c1W1, c1b1, c1W2, c1b2, c2W, c2b,
                                   ws_cenf, ws_cnorm);
  k_simi<<<(B_ * N_) / 256, 256, 0, stream>>>(x, i1W1, i1b1, i1W2, i1b2, i2W,
                                              i2b, ws_cenf, ws_cnorm, ws_simi);
  k_gcn<<<512, 256, 0, stream>>>(inp, adj, gW, gB, ws_simi, out);
  k_zero<<<1, 512, 0, stream>>>(ws_acc);
  k_reduce<<<(B_ * N_) / 16, 256, 0, stream>>>(out, ws_acc);
  k_final<<<1, 512, 0, stream>>>(cen, ws_acc, out + (size_t)B_ * N_ * C_ * S_);
}
